// LocalMSA_14370960572445
// MI455X (gfx1250) — matrix operand order repaired and checked by reading, not compiled, not run on hardware
//
#include <hip/hip_runtime.h>

// ---------------------------------------------------------------------------
// LocalMSA for MI455X (gfx1250, wave32, WMMA).
// B=16 G=64 L=256 C=128 H=2 c=64.  One block per (b,g,h): 2048 blocks x 256 thr.
// All GEMMs via v_wmma_f32_16x16x32_bf16 (fp32 accumulate).
// ---------------------------------------------------------------------------

typedef __attribute__((ext_vector_type(16))) __bf16 v16bf;
typedef __attribute__((ext_vector_type(8)))  float  v8f;

union Frag {
    v16bf        v;
    unsigned short u16[16];
    unsigned int   u32[8];
};

static __device__ __forceinline__ unsigned short f2bf(float f) {
    // round-to-nearest-even fp32 -> bf16
    unsigned int u = __float_as_uint(f);
    u += 0x7FFFu + ((u >> 16) & 1u);
    return (unsigned short)(u >> 16);
}

static __device__ __forceinline__ void load8h(const unsigned short* p, unsigned int* dst) {
    // 8 contiguous bf16 (16B) as 4 dword LDS loads; p is 4B aligned by construction
    const unsigned int* q = (const unsigned int*)p;
    dst[0] = q[0]; dst[1] = q[1]; dst[2] = q[2]; dst[3] = q[3];
}

static __device__ __forceinline__ v8f wmma_bf16(const Frag& a, const Frag& b, v8f c) {
    return __builtin_amdgcn_wmma_f32_16x16x32_bf16(
        /*neg_a=*/false, a.v, /*neg_b=*/false, b.v,
        /*c_mod=*/(short)0, c, /*reuse_a=*/false, /*reuse_b=*/false);
}

// LDS row strides (in bf16 elements) — padded for bank spread + 4B alignment.
#define LQ  68      // q/k rows:  [256][68]
#define LVT 260     // vT rows:   [64][260]
#define LW  132     // W rows:    [192][132]
#define LP  34      // P rows:    [16][34] per row-tile, per wave
#define SMEM_BYTES ((256*LQ + 256*LQ + 64*LVT + 192*LW) * 2)   // 153600 B

__global__ __launch_bounds__(256, 1)
void msa_kernel(const float* __restrict__ x, const float* __restrict__ Wq,
                const float* __restrict__ bq, float* __restrict__ out)
{
    extern __shared__ char smem[];
    unsigned short* qs  = (unsigned short*)smem;     // Q  bf16 [256][LQ]
    unsigned short* ks  = qs  + 256 * LQ;            // K  bf16 [256][LQ]
    unsigned short* vts = ks  + 256 * LQ;            // V^T bf16 [64][LVT]
    unsigned short* ws  = vts + 64  * LVT;           // W slice bf16 [192][LW]
    unsigned short* ps  = ws;                        // P scratch aliases W (phase 2)

    const int tid  = threadIdx.x;
    const int lane = tid & 31;
    const int wave = tid >> 5;
    const int ln16 = lane & 15;
    const int hi   = lane >> 4;          // half-wave select (ISA fragment swizzle)
    const int h    = blockIdx.x & 1;     // head
    const int bg   = blockIdx.x >> 1;    // (b,g) flat

    const float* xg = x   + (size_t)bg * (256 * 128);
    float*       og = out + (size_t)bg * (256 * 128);

    // ---- stage this head's W slice (192 x 128) into LDS as bf16 ----
    for (int idx = tid; idx < 192 * 128; idx += 256) {
        int s     = idx >> 7;            // slice row 0..191 (q|k|v x 64)
        int ch    = idx & 127;
        int which = s >> 6;
        int j     = s & 63;
        int d     = which * 128 + h * 64 + j;   // global row of W_qkv
        ws[s * LW + ch] = f2bf(Wq[d * 128 + ch]);
    }
    __syncthreads();

    // ---- Phase 1: qkv projection. Wave owns row strips 2w, 2w+1 (32 rows). ----
    // A-fragments of x (bf16, contiguous: lane<16 K=0..15, lane>=16 K=16..31)
    Frag afr[2][4];
    #pragma unroll
    for (int rt = 0; rt < 2; ++rt) {
        int row = (wave * 2 + rt) * 16 + ln16;
        const float* xr = xg + row * 128;
        #pragma unroll
        for (int t = 0; t < 4; ++t) {
            int c0 = t * 32 + hi * 16;
            #pragma unroll
            for (int e = 0; e < 8; ++e) {
                afr[rt][t].u16[e]     = f2bf(xr[c0 + e]);
                afr[rt][t].u16[e + 8] = f2bf(xr[c0 + 8 + e]);
            }
        }
    }

    for (int n = 0; n < 12; ++n) {               // 12 output col-tiles (q0..3 k0..3 v0..3)
        int which = n >> 2;
        int ncol  = (n & 3) * 16;
        // B-fragments from W LDS: lane holds column N=ln16, 16 contiguous K per half
        Frag bfr[4];
        {
            const unsigned short* wr = ws + (n * 16 + ln16) * LW;
            #pragma unroll
            for (int t = 0; t < 4; ++t) {
                const unsigned short* p = wr + t * 32 + hi * 16;
                load8h(p,     &bfr[t].u32[0]);
                load8h(p + 8, &bfr[t].u32[4]);
            }
        }
        float bias = bq[which * 128 + h * 64 + ncol + ln16];

        #pragma unroll
        for (int rt = 0; rt < 2; ++rt) {
            v8f c;
            #pragma unroll
            for (int r = 0; r < 8; ++r) c[r] = bias;
            #pragma unroll
            for (int t = 0; t < 4; ++t) c = wmma_bf16(afr[rt][t], bfr[t], c);
            // scatter C tile (f32 layout: VGPR r -> row r / r+8) to LDS as bf16
            int rowBase = (wave * 2 + rt) * 16;
            #pragma unroll
            for (int r = 0; r < 8; ++r) {
                int row = rowBase + r + hi * 8;
                int col = ncol + ln16;
                unsigned short bv = f2bf(c[r]);
                if (which == 0)      qs[row * LQ + col]   = bv;
                else if (which == 1) ks[row * LQ + col]   = bv;
                else                 vts[col * LVT + row] = bv;   // V stored transposed
            }
        }
    }
    __syncthreads();   // qkv visible to all waves; W region now dead -> P scratch

    // ---- Phase 2: flash attention over the wave's 32 query rows ----
    Frag qfr[2][2];
    #pragma unroll
    for (int rt = 0; rt < 2; ++rt) {
        const unsigned short* qr = qs + ((wave * 2 + rt) * 16 + ln16) * LQ;
        #pragma unroll
        for (int t = 0; t < 2; ++t) {           // K=64 -> 2 wmma steps
            int c0 = t * 32 + hi * 16;          // contiguous K per half
            load8h(qr + c0,      &qfr[rt][t].u32[0]);
            load8h(qr + c0 + 8,  &qfr[rt][t].u32[4]);
        }
    }

    float mrun[2][8], lrun[2][8];
    v8f acc[2][4];
    #pragma unroll
    for (int rt = 0; rt < 2; ++rt) {
        #pragma unroll
        for (int r = 0; r < 8; ++r) { mrun[rt][r] = -1e30f; lrun[rt][r] = 0.0f; }
        #pragma unroll
        for (int ct = 0; ct < 4; ++ct) {
            v8f z = {0.f,0.f,0.f,0.f,0.f,0.f,0.f,0.f};
            acc[rt][ct] = z;
        }
    }

    const float inv_scale = 0.088388347648318447f;   // 1/sqrt(128)

    for (int jp = 0; jp < 8; ++jp) {                 // 32 keys per step
        // K B-fragments for the two 16-key tiles (lane = key column, contiguous ch)
        Frag kf[2][2];
        #pragma unroll
        for (int jj = 0; jj < 2; ++jj) {
            const unsigned short* kr = ks + ((jp * 2 + jj) * 16 + ln16) * LQ;
            #pragma unroll
            for (int t = 0; t < 2; ++t) {
                const unsigned short* p = kr + t * 32 + hi * 16;
                load8h(p,     &kf[jj][t].u32[0]);
                load8h(p + 8, &kf[jj][t].u32[4]);
            }
        }

        #pragma unroll
        for (int rt = 0; rt < 2; ++rt) {
            v8f s0 = {0.f,0.f,0.f,0.f,0.f,0.f,0.f,0.f};
            v8f s1 = {0.f,0.f,0.f,0.f,0.f,0.f,0.f,0.f};
            s0 = wmma_bf16(qfr[rt][0], kf[0][0], s0);
            s0 = wmma_bf16(qfr[rt][1], kf[0][1], s0);
            s1 = wmma_bf16(qfr[rt][0], kf[1][0], s1);
            s1 = wmma_bf16(qfr[rt][1], kf[1][1], s1);

            unsigned short* prow = ps + (wave * 2 + rt) * (16 * LP);
            #pragma unroll
            for (int r = 0; r < 8; ++r) {
                float a0 = s0[r] * inv_scale;
                float a1 = s1[r] * inv_scale;
                // row max over 32 keys: xor masks 1..8 stay inside 16-lane halves,
                // matching C-layout rows (VGPR r -> row r or r+8 per half)
                float tm = fmaxf(a0, a1);
                tm = fmaxf(tm, __shfl_xor(tm, 1, 32));
                tm = fmaxf(tm, __shfl_xor(tm, 2, 32));
                tm = fmaxf(tm, __shfl_xor(tm, 4, 32));
                tm = fmaxf(tm, __shfl_xor(tm, 8, 32));
                float mnew = fmaxf(mrun[rt][r], tm);
                float fac  = __expf(mrun[rt][r] - mnew);
                float p0   = __expf(a0 - mnew);
                float p1   = __expf(a1 - mnew);
                float rs = p0 + p1;
                rs += __shfl_xor(rs, 1, 32);
                rs += __shfl_xor(rs, 2, 32);
                rs += __shfl_xor(rs, 4, 32);
                rs += __shfl_xor(rs, 8, 32);
                lrun[rt][r] = lrun[rt][r] * fac + rs;
                mrun[rt][r] = mnew;
                #pragma unroll
                for (int ct = 0; ct < 4; ++ct) acc[rt][ct][r] *= fac;
                int row = r + hi * 8;
                prow[row * LP + ln16]      = f2bf(p0);
                prow[row * LP + 16 + ln16] = f2bf(p1);
            }
        }

        // P @ V : per-wave P scratch (LDS ops are in-order within a wave)
        Frag pf[2];
        #pragma unroll
        for (int rt = 0; rt < 2; ++rt) {
            const unsigned short* pr = ps + (wave * 2 + rt) * (16 * LP) + ln16 * LP;
            int c0 = hi * 16;                                  // contiguous K per half
            load8h(pr + c0,      &pf[rt].u32[0]);
            load8h(pr + c0 + 8,  &pf[rt].u32[4]);
        }
        #pragma unroll
        for (int ct = 0; ct < 4; ++ct) {
            Frag vf;
            const unsigned short* vr = vts + (ct * 16 + ln16) * LVT + jp * 32 + hi * 16;
            load8h(vr,     &vf.u32[0]);
            load8h(vr + 8, &vf.u32[4]);
            #pragma unroll
            for (int rt = 0; rt < 2; ++rt)
                acc[rt][ct] = wmma_bf16(pf[rt], vf, acc[rt][ct]);
        }
    }

    // ---- epilogue: divide by softmax denom, store fp32 out[b,g,l, h*64+c] ----
    #pragma unroll
    for (int rt = 0; rt < 2; ++rt) {
        int rowBase = (wave * 2 + rt) * 16;
        #pragma unroll
        for (int r = 0; r < 8; ++r) {
            float invl = 1.0f / lrun[rt][r];
            int row = rowBase + r + hi * 8;
            float* orow = og + row * 128 + h * 64;
            #pragma unroll
            for (int ct = 0; ct < 4; ++ct)
                orow[ct * 16 + ln16] = acc[rt][ct][r] * invl;
        }
    }
}

extern "C" void kernel_launch(void* const* d_in, const int* in_sizes, int n_in,
                              void* d_out, int out_size, void* d_ws, size_t ws_size,
                              hipStream_t stream) {
    (void)in_sizes; (void)n_in; (void)out_size; (void)d_ws; (void)ws_size;
    const float* x = (const float*)d_in[0];
    const float* W = (const float*)d_in[1];
    const float* b = (const float*)d_in[2];
    float* out = (float*)d_out;
    dim3 grid(16 * 64 * 2);   // (b,g) x heads
    dim3 block(256);          // 8 wave32
    hipLaunchKernelGGL(msa_kernel, grid, block, SMEM_BYTES, stream, x, W, b, out);
}
